// TinyMambaANC_52243982188792
// MI455X (gfx1250) — matrix operand order repaired and checked by reading, not compiled, not run on hardware
//
#include <hip/hip_runtime.h>
#include <hip/hip_bf16.h>
#include <math.h>

// ---------------------------------------------------------------------------
// TinyMamba fused pipeline for MI455X (gfx1250, wave32, WMMA)
//   enc conv+gelu -> [LN -> in_proj -> silu -> Bproj -> decay-scan ->
//                     Cproj+skip -> gate -> out_proj -> residual] x2 -> dec conv
// Scan parallelized via halo-recompute: 0.9^192 ~ 1.7e-9 << fp32 eps, so each
// 512-token chunk is independent given a 192-token warmup halo. One chunk per
// wave32; all GEMMs are v_wmma_f32_16x16x32_f16 on 16-token tiles.
// This revision replaces IEEE divide/rsqrt chains in silu/layernorm with
// single-instruction v_rcp_f32 / v_rsq_f32 approximations.
// ---------------------------------------------------------------------------

#define BATCH    16
#define T_FULL   131072
#define T2       65536
#define DM       32
#define DI       32
#define DSTATE   16
#define MAIN     512
#define HALO     192
#define CHUNKS_PER_BATCH (T2 / MAIN)      // 128
#define WAVES_PER_BLOCK  8

typedef __attribute__((ext_vector_type(16))) _Float16 v16h;
typedef __attribute__((ext_vector_type(8)))  _Float16 v8h;
typedef __attribute__((ext_vector_type(8)))  float    v8f;

// fast activations: one v_exp_f32 + one v_rcp_f32 instead of full div chain
__device__ __forceinline__ float siluf(float x) {
    return x * __builtin_amdgcn_rcpf(1.0f + __expf(-x));
}
__device__ __forceinline__ float geluf(float x) {
    return 0.5f * x * (1.0f + erff(x * 0.70710678118654752f));
}

// A fragment (16x32 f16, M x K) from LDS tile stored row-major [m][k].
// ISA 7.12.2: lanes 0-15 row m=lane, halves 0..7 = K0..7, 8..15 = K16..23;
//             lanes 16-31 row m=lane-16, halves 0..7 = K8..15, 8..15 = K24..31.
__device__ __forceinline__ v16h ldsA(const _Float16* a, int lane) {
    int row = lane & 15;
    int hi  = (lane >> 4) & 1;
    const _Float16* p = a + row * 32 + hi * 8;
    v8h lo = *(const v8h*)(p);
    v8h hv = *(const v8h*)(p + 16);
    v16h r;
#pragma unroll
    for (int i = 0; i < 8; ++i) { r[i] = lo[i]; r[i + 8] = hv[i]; }
    return r;
}

// B fragment (32x16 f16, K x N) from LDS tile stored column-major [n][k]:
// lanes 0-15 col n=lane with K0..7,K16..23; lanes 16-31 col n=lane-16 with K8..15,K24..31.
__device__ __forceinline__ v16h ldsB(const _Float16* b, int lane) {
    int col = lane & 15;
    int hi  = (lane >> 4) & 1;
    const _Float16* p = b + col * 32 + hi * 8;
    v8h lo = *(const v8h*)(p);
    v8h hv = *(const v8h*)(p + 16);
    v16h r;
#pragma unroll
    for (int i = 0; i < 8; ++i) { r[i] = lo[i]; r[i + 8] = hv[i]; }
    return r;
}

__device__ __forceinline__ v8f wmma_f16(v16h a, v16h b, v8f c) {
    return __builtin_amdgcn_wmma_f32_16x16x32_f16(false, a, false, b, (short)0, c, false, false);
}

// ---------------------------------------------------------------------------
// Encoder: h[b][t][c] = gelu( sum_k x[b][2t+k-1] * w[c][k] + bias[c] )
// ---------------------------------------------------------------------------
__global__ __launch_bounds__(256) void tm_enc_kernel(
    const float* __restrict__ x, const float* __restrict__ w,
    const float* __restrict__ bias, float* __restrict__ h)
{
    size_t gid = (size_t)blockIdx.x * 256 + threadIdx.x;   // b*T2*32 + t*32 + c
    int c = (int)(gid & 31);
    size_t bt = gid >> 5;
    int t = (int)(bt % T2);
    int b = (int)(bt / T2);
    const float* xb = x + (size_t)b * T_FULL;
    float acc = bias[c];
#pragma unroll
    for (int k = 0; k < 4; ++k) {
        int j = 2 * t + k - 1;
        float xv = (j >= 0 && j < T_FULL) ? xb[j] : 0.0f;
        acc += xv * w[c * 4 + k];
    }
    h[gid] = geluf(acc);
}

// ---------------------------------------------------------------------------
// Fused SSM layer: one 512-token chunk (+192 halo) per wave32.
// ---------------------------------------------------------------------------
__global__ __launch_bounds__(256) void tm_layer_kernel(
    const float* __restrict__ hin, float* __restrict__ hout,
    const float* __restrict__ g_lng, const float* __restrict__ g_lnb,
    const float* __restrict__ g_wi,  const float* __restrict__ g_bi,
    const float* __restrict__ g_wb,  const float* __restrict__ g_wc,
    const float* __restrict__ g_dp,  const float* __restrict__ g_wo,
    const float* __restrict__ g_bo,  int layer)
{
    // Weights in f16 B-tile layout (column-major [n][k], 32 halfs per column)
    __shared__ __align__(16) _Float16 Wi[64 * 32];   // in_proj: 4 tiles of 16 cols
    __shared__ __align__(16) _Float16 Wb[16 * 32];   // B-proj: 1 tile
    __shared__ __align__(16) _Float16 Wc[32 * 32];   // C-proj: 2 tiles, K padded 16->32 w/ zeros
    __shared__ __align__(16) _Float16 Wo[32 * 32];   // out_proj: 2 tiles
    __shared__ float s_lng[32], s_lnb[32], s_bi[64], s_dp[32], s_bo[32];
    // Per-wave scratch tiles
    __shared__ __align__(16) _Float16 A1[WAVES_PER_BLOCK][512]; // LN'd input (16x32)
    __shared__ __align__(16) _Float16 A2[WAVES_PER_BLOCK][512]; // silu(x_proj)
    __shared__ __align__(16) _Float16 A3[WAVES_PER_BLOCK][512]; // scan states, K-padded
    __shared__ __align__(16) _Float16 A4[WAVES_PER_BLOCK][512]; // gated y
    __shared__ float BS[WAVES_PER_BLOCK][256];                  // b_seq (16 tok x 16 st)

    int tid = threadIdx.x;
    for (int i = tid; i < 64 * 32; i += 256) Wi[i] = (_Float16)g_wi[layer * 64 * 32 + i];
    for (int i = tid; i < 16 * 32; i += 256) Wb[i] = (_Float16)g_wb[layer * 16 * 32 + i];
    for (int i = tid; i < 32 * 32; i += 256) {
        int n = i >> 5, k = i & 31;
        Wc[i] = (k < 16) ? (_Float16)g_wc[layer * 32 * 16 + n * 16 + k] : (_Float16)0.0f;
    }
    for (int i = tid; i < 32 * 32; i += 256) Wo[i] = (_Float16)g_wo[layer * 32 * 32 + i];
    if (tid < 32) {
        s_lng[tid] = g_lng[layer * 32 + tid];
        s_lnb[tid] = g_lnb[layer * 32 + tid];
        s_dp[tid]  = g_dp[layer * 32 + tid];
        s_bo[tid]  = g_bo[layer * 32 + tid];
    }
    if (tid < 64) s_bi[tid] = g_bi[layer * 64 + tid];
    __syncthreads();

    int wave  = tid >> 5;
    int lane  = tid & 31;
    int chunk = blockIdx.x * WAVES_PER_BLOCK + wave;
    int batch = chunk / CHUNKS_PER_BATCH;
    int t0    = (chunk % CHUNKS_PER_BATCH) * MAIN;
    const float* inB = hin + (size_t)batch * T2 * DM;
    float*      outB = hout + (size_t)batch * T2 * DM;

    _Float16* a1 = A1[wave];
    _Float16* a2 = A2[wave];
    _Float16* a3 = A3[wave];
    _Float16* a4 = A4[wave];
    float*    bs = BS[wave];

    // zero the K=16..31 padding columns of the state tile once
    for (int i = lane; i < 256; i += 32) a3[(i >> 4) * 32 + 16 + (i & 15)] = (_Float16)0.0f;

    float hstate = 0.0f;            // lane (l&15) carries state s = l&15
    const int NTILE = (HALO + MAIN) / 16;
    const int n = lane & 15;        // output-column / feature index within tile
    v8f zero = {};

#pragma unroll 1
    for (int tile = 0; tile < NTILE; ++tile) {
        int tstart = t0 - HALO + tile * 16;
        if (tstart < 0) continue;            // sequence start: h0 == 0, no halo needed
        bool halo = tstart < t0;

        __builtin_prefetch(inB + (size_t)(tstart + 16) * DM + lane, 0, 1);

        // ---- LayerNorm over 32 features (lane = feature), 16 tokens ----
#pragma unroll 1
        for (int t = 0; t < 16; ++t) {
            float v = inB[(size_t)(tstart + t) * DM + lane];
            float s = v;
            for (int off = 16; off; off >>= 1) s += __shfl_xor(s, off, 32);
            float mu = s * (1.0f / 32.0f);
            float d = v - mu;
            float q = d * d;
            for (int off = 16; off; off >>= 1) q += __shfl_xor(q, off, 32);
            float inv = __builtin_amdgcn_rsqf(q * (1.0f / 32.0f) + 1e-5f);
            a1[t * 32 + lane] = (_Float16)(d * inv * s_lng[lane] + s_lnb[lane]);
        }

        // ---- in_proj (x half) + SiLU ----
        v16h fa  = ldsA(a1, lane);
        v8f  xp0 = wmma_f16(fa, ldsB(Wi + 0,   lane), zero);  // feats 0..15
        v8f  xp1 = wmma_f16(fa, ldsB(Wi + 512, lane), zero);  // feats 16..31
#pragma unroll
        for (int r = 0; r < 8; ++r) {
            int m = (lane < 16) ? r : r + 8;                  // token row (C layout)
            float u0 = siluf(xp0[r] + s_bi[n]);
            float u1 = siluf(xp1[r] + s_bi[16 + n]);
            xp0[r] = u0; xp1[r] = u1;
            a2[m * 32 + n]      = (_Float16)u0;
            a2[m * 32 + 16 + n] = (_Float16)u1;
        }

        // ---- B projection -> b_seq (16 tok x 16 states) ----
        v8f bacc = wmma_f16(ldsA(a2, lane), ldsB(Wb, lane), zero);
#pragma unroll
        for (int r = 0; r < 8; ++r) {
            int m = (lane < 16) ? r : r + 8;
            bs[m * 16 + n] = bacc[r];
        }

        // ---- sequential in-tile scan: h = 0.9h + 0.1b (lane = state) ----
#pragma unroll 1
        for (int t = 0; t < 16; ++t) {
            float bv = bs[t * 16 + n];
            hstate = 0.9f * hstate + 0.1f * bv;
            if (lane < 16) a3[t * 32 + lane] = (_Float16)hstate;
        }

        if (halo) continue;                  // halo: only state warmup

        // ---- in_proj (z half) ----
        v8f z0 = wmma_f16(fa, ldsB(Wi + 1024, lane), zero);
        v8f z1 = wmma_f16(fa, ldsB(Wi + 1536, lane), zero);

        // ---- C projection, accumulator seeded with D * x_proj ----
        v8f cy0, cy1;
#pragma unroll
        for (int r = 0; r < 8; ++r) { cy0[r] = xp0[r] * s_dp[n]; cy1[r] = xp1[r] * s_dp[16 + n]; }
        v16h ha = ldsA(a3, lane);
        v8f y0 = wmma_f16(ha, ldsB(Wc, lane),       cy0);
        v8f y1 = wmma_f16(ha, ldsB(Wc + 512, lane), cy1);

        // ---- gate with silu(z), stage for out_proj ----
#pragma unroll
        for (int r = 0; r < 8; ++r) {
            int m = (lane < 16) ? r : r + 8;
            float g0 = siluf(z0[r] + s_bi[32 + n]);
            float g1 = siluf(z1[r] + s_bi[48 + n]);
            float w0 = y0[r] * g0;
            float w1 = y1[r] * g1;
            a4[m * 32 + n]      = (_Float16)w0;
            a4[m * 32 + 16 + n] = (_Float16)w1;
        }

        // ---- out_proj + bias + residual ----
        v16h ya = ldsA(a4, lane);
        v8f o0 = wmma_f16(ya, ldsB(Wo, lane),       zero);
        v8f o1 = wmma_f16(ya, ldsB(Wo + 512, lane), zero);
#pragma unroll
        for (int r = 0; r < 8; ++r) {
            int m = (lane < 16) ? r : r + 8;
            size_t gi = (size_t)(tstart + m) * DM;
            outB[gi + n]      = inB[gi + n]      + o0[r] + s_bo[n];
            outB[gi + 16 + n] = inB[gi + 16 + n] + o1[r] + s_bo[16 + n];
        }
    }
}

// ---------------------------------------------------------------------------
// Decoder: stride-2 transposed conv (k=4, pad (2,2)), 32ch -> 1ch
// t even: taps h[(t-2)/2]*w[c][3] + h[t/2]*w[c][1]
// t odd : taps h[(t-1)/2]*w[c][2] + h[(t+1)/2]*w[c][0]
// ---------------------------------------------------------------------------
__global__ __launch_bounds__(256) void tm_dec_kernel(
    const float* __restrict__ h, const float* __restrict__ w,
    const float* __restrict__ bias, float* __restrict__ out)
{
    size_t gid = (size_t)blockIdx.x * 256 + threadIdx.x;   // b*T_FULL + t
    int t = (int)(gid % T_FULL);
    int b = (int)(gid / T_FULL);
    const float* hb = h + (size_t)b * T2 * DM;
    float acc = bias[0];
    int i0, i1, k0, k1;
    bool v0, v1;
    if ((t & 1) == 0) { i0 = (t - 2) / 2; k0 = 3; i1 = t / 2;       k1 = 1; v0 = (t >= 2); v1 = true; }
    else              { i0 = (t - 1) / 2; k0 = 2; i1 = (t + 1) / 2; k1 = 0; v0 = true;     v1 = (i1 < T2); }
#pragma unroll 4
    for (int c = 0; c < 32; ++c) {
        if (v0) acc += w[c * 4 + k0] * hb[(size_t)i0 * DM + c];
        if (v1) acc += w[c * 4 + k1] * hb[(size_t)i1 * DM + c];
    }
    out[gid] = acc;
}

// ---------------------------------------------------------------------------
extern "C" void kernel_launch(void* const* d_in, const int* in_sizes, int n_in,
                              void* d_out, int out_size, void* d_ws, size_t ws_size,
                              hipStream_t stream)
{
    const float* x        = (const float*)d_in[0];
    const float* enc_w    = (const float*)d_in[1];
    const float* enc_b    = (const float*)d_in[2];
    const float* ln_g     = (const float*)d_in[3];
    const float* ln_b     = (const float*)d_in[4];
    const float* inproj_w = (const float*)d_in[5];
    const float* inproj_b = (const float*)d_in[6];
    const float* B_w      = (const float*)d_in[7];
    const float* C_w      = (const float*)d_in[8];
    const float* D_p      = (const float*)d_in[9];
    const float* outproj_w= (const float*)d_in[10];
    const float* outproj_b= (const float*)d_in[11];
    const float* dec_w    = (const float*)d_in[12];
    const float* dec_b    = (const float*)d_in[13];

    float* h0 = (float*)d_ws;
    float* h1 = h0 + (size_t)BATCH * T2 * DM;   // ping-pong hidden buffers (128MB each)

    // Encoder: 16*65536*32 threads
    tm_enc_kernel<<<(BATCH * T2 * DM) / 256, 256, 0, stream>>>(x, enc_w, enc_b, h0);

    // Two fused SSM layers: 2048 chunks, 8 wave32 chunks per block
    dim3 lgrid((BATCH * CHUNKS_PER_BATCH) / WAVES_PER_BLOCK);   // 256 blocks
    tm_layer_kernel<<<lgrid, 256, 0, stream>>>(h0, h1, ln_g, ln_b, inproj_w, inproj_b,
                                               B_w, C_w, D_p, outproj_w, outproj_b, 0);
    tm_layer_kernel<<<lgrid, 256, 0, stream>>>(h1, h0, ln_g, ln_b, inproj_w, inproj_b,
                                               B_w, C_w, D_p, outproj_w, outproj_b, 1);

    // Decoder: 16*131072 threads
    tm_dec_kernel<<<(BATCH * T_FULL) / 256, 256, 0, stream>>>(h0, dec_w, dec_b, (float*)d_out);
}
